// QMixer_27204322853102
// MI455X (gfx1250) — compile-verified, matrix operand order checked by reading, not run on hardware
//
#include <hip/hip_runtime.h>
#include <hip/hip_bf16.h>

// Problem constants (from reference)
constexpr int BSZ  = 4096;
constexpr int NA   = 128;   // agents
constexpr int NOBS = 128;
constexpr int NACT = 8;
constexpr int SD   = 4096;  // state dim
constexpr int NE   = 32;    // embedding
constexpr int NH1  = 128;
constexpr int NH2  = 64;

typedef __attribute__((ext_vector_type(8)))  __bf16 v8bf;
typedef __attribute__((ext_vector_type(16))) __bf16 v16bf;
typedef __attribute__((ext_vector_type(8)))  float  v8f;
typedef unsigned int v4u __attribute__((ext_vector_type(4)));
typedef int v8i __attribute__((ext_vector_type(8)));
typedef int v4i __attribute__((ext_vector_type(4)));

// ---------------------------------------------------------------------------
// CDNA5 WMMA helpers (wave32). 16x16x32 bf16, f32 accumulate.
// A-frag layout (ISA 7.12.2): lane L holds row (L&15); lanes<16 carry
// K={0..7,16..23}, lanes>=16 carry K={8..15,24..31} -> two 16B LDS loads.
// B is staged transposed (N x K) in LDS so a B-frag loads identically.
// ---------------------------------------------------------------------------
__device__ __forceinline__ v16bf lds_frag(const __bf16* rowPtr, int half) {
  const v8bf* p = (const v8bf*)rowPtr;     // rowPtr is 16B aligned
  v8bf lo = p[half];                       // K offset: half*8
  v8bf hi = p[2 + half];                   // K offset: 16 + half*8
  v16bf r;
#pragma unroll
  for (int i = 0; i < 8; ++i) { r[i] = lo[i]; r[8 + i] = hi[i]; }
  return r;
}

// A-frag from an f32 LDS slab (TDM-staged raw): convert on read. The
// v_cvt ops co-execute with the WMMA pipe (XDL-as-TRANS tracking).
__device__ __forceinline__ v16bf lds_frag_f32(const float* rowPtr, int half) {
  const float* p0 = rowPtr + half * 8;
  const float* p1 = rowPtr + 16 + half * 8;
  v16bf r;
#pragma unroll
  for (int i = 0; i < 8; ++i) { r[i] = (__bf16)p0[i]; r[8 + i] = (__bf16)p1[i]; }
  return r;
}

__device__ __forceinline__ v8f wmma_bf16(v16bf a, v16bf b, v8f c) {
  return __builtin_amdgcn_wmma_f32_16x16x32_bf16(false, a, false, b,
                                                 (short)0, c, false, false);
}

// ---------------------------------------------------------------------------
// Tensor Data Mover: DMA a 2-D f32 tile [rows x cols] (row-major, element
// stride `srcStride` between rows) from global memory into LDS (contiguous).
// D# packing per CDNA5 ISA 8.3/8.4: group0 = {count=1, lds_addr,
// global_addr, type=2}; group1 = {mask=0, data_size=4B, tensor dims,
// tile dims, tensor_dim0_stride}. Groups 2/3 zero (2-D tensor).
// This toolchain's builtin takes 6 args (g0, g1, g2, g3, extra, cpol).
// ---------------------------------------------------------------------------
__device__ __forceinline__ void tdm_load_2d_f32(unsigned ldsAddr,
                                                const float* gsrc,
                                                unsigned cols, unsigned rows,
                                                unsigned long srcStride) {
  unsigned long ga = (unsigned long)(const void*)gsrc;
  v4u g0;
  g0[0] = 1u;                                           // count=1, user mode
  g0[1] = ldsAddr;                                      // LDS byte address
  g0[2] = (unsigned)(ga & 0xffffffffu);                 // global_addr[31:0]
  g0[3] = (unsigned)((ga >> 32) & 0x1ffffffu) | (2u << 30);  // addr[56:32], type=2
  v8i g1;
  g1[0] = (int)(2u << 16);                              // data_size = 4B
  g1[1] = (int)((cols & 0xffffu) << 16);                // tensor_dim0[15:0]
  g1[2] = (int)((cols >> 16) | ((rows & 0xffffu) << 16));   // d0 hi | d1 lo
  g1[3] = (int)((rows >> 16) | ((cols & 0xffffu) << 16));   // d1 hi | tile_dim0
  g1[4] = (int)(rows & 0xffffu);                        // tile_dim1 (tile_dim2=0)
  g1[5] = (int)(unsigned)(srcStride & 0xffffffffu);     // dim0_stride[31:0]
  g1[6] = (int)(unsigned)((srcStride >> 32) & 0xffffu); // dim0_stride[47:32]
  g1[7] = 0;
  v4i gz4 = {0, 0, 0, 0};
  v8i gz8 = {0, 0, 0, 0, 0, 0, 0, 0};
  __builtin_amdgcn_tensor_load_to_lds(g0, g1, gz4, gz4, gz8, 0);
}

// ---------------------------------------------------------------------------
// Kernel 1: per-agent 3-layer MLP, fused with row max / argmax.
// grid = (BSZ/128, NA), block = 256 threads (8 waves). Each wave owns a
// 16-row strip of the 128-row batch tile; all agent weights live in LDS.
// ---------------------------------------------------------------------------
__global__ __launch_bounds__(256) void agent_mlp_kernel(
    const float* __restrict__ obs,
    const float* __restrict__ W1, const float* __restrict__ b1,
    const float* __restrict__ W2, const float* __restrict__ b2,
    const float* __restrict__ W3, const float* __restrict__ b3,
    float* __restrict__ qs, int* __restrict__ actions)
{
  __shared__ __align__(16) __bf16 sObs[128 * 128];  // batch tile, M x K
  __shared__ __align__(16) __bf16 sW1t[128 * 128];  // W1^T, N x K
  __shared__ __align__(16) __bf16 sW2t[64 * 128];   // W2^T
  __shared__ __align__(16) __bf16 sW3t[16 * 64];    // W3^T (cols 8..15 zero)
  __shared__ __align__(16) __bf16 sH1[128 * 128];   // layer-1 activations
  __shared__ __align__(16) __bf16 sH2[128 * 64];    // layer-2 activations
  __shared__ __align__(16) float  sQ[128 * 16];     // logits (padded to 16)

  const int tid  = threadIdx.x;
  const int lane = tid & 31;
  const int wv   = tid >> 5;
  const int half = lane >> 4;
  const int nl   = lane & 15;
  const int b0   = blockIdx.x * 128;
  const int a    = blockIdx.y;

  // ---- stage obs tile (coalesced float4 reads, bf16 convert) ----
  {
    const float* src = obs + ((size_t)b0 * NA + a) * NOBS;
    for (int idx = tid; idx < 128 * 128 / 4; idx += 256) {
      int r  = idx >> 5;            // 32 float4 per 128-wide row
      int c4 = (idx & 31) * 4;
      float4 v = *(const float4*)(src + (size_t)r * NA * NOBS + c4);
      __bf16* d = &sObs[r * 128 + c4];
      d[0] = (__bf16)v.x; d[1] = (__bf16)v.y;
      d[2] = (__bf16)v.z; d[3] = (__bf16)v.w;
    }
    const float* w1p = W1 + (size_t)a * NOBS * NH1;
    for (int idx = tid; idx < 128 * 128; idx += 256) {
      int k = idx >> 7, n = idx & 127;
      sW1t[n * 128 + k] = (__bf16)w1p[k * NH1 + n];     // transpose-stage
    }
    const float* w2p = W2 + (size_t)a * NH1 * NH2;
    for (int idx = tid; idx < 128 * 64; idx += 256) {
      int k = idx >> 6, n = idx & 63;
      sW2t[n * 128 + k] = (__bf16)w2p[k * NH2 + n];
    }
    const float* w3p = W3 + (size_t)a * NH2 * NACT;
    for (int idx = tid; idx < 16 * 64; idx += 256) {
      int k = idx >> 4, n = idx & 15;
      sW3t[n * 64 + k] = (n < NACT) ? (__bf16)w3p[k * NACT + n] : (__bf16)0.0f;
    }
  }
  __syncthreads();

  const int m0 = wv * 16;

  // ---- layer 1: h1 = relu(obs @ W1 + b1) : wave computes 16 x 128 ----
  {
    v8f acc[8] = {};
#pragma unroll
    for (int kk = 0; kk < 128; kk += 32) {
      v16bf af = lds_frag(&sObs[(m0 + nl) * 128 + kk], half);
#pragma unroll
      for (int t = 0; t < 8; ++t) {
        v16bf bf = lds_frag(&sW1t[(t * 16 + nl) * 128 + kk], half);
        acc[t] = wmma_bf16(af, bf, acc[t]);
      }
    }
#pragma unroll
    for (int t = 0; t < 8; ++t) {
      float bias = b1[a * NH1 + t * 16 + nl];
#pragma unroll
      for (int v = 0; v < 8; ++v) {
        int m = v + half * 8;                      // C layout row
        float val = fmaxf(acc[t][v] + bias, 0.0f);
        sH1[(m0 + m) * 128 + t * 16 + nl] = (__bf16)val;
      }
    }
  }
  __syncthreads();

  // ---- layer 2: h2 = relu(h1 @ W2 + b2) : 16 x 64 ----
  {
    v8f acc[4] = {};
#pragma unroll
    for (int kk = 0; kk < 128; kk += 32) {
      v16bf af = lds_frag(&sH1[(m0 + nl) * 128 + kk], half);
#pragma unroll
      for (int t = 0; t < 4; ++t) {
        v16bf bf = lds_frag(&sW2t[(t * 16 + nl) * 128 + kk], half);
        acc[t] = wmma_bf16(af, bf, acc[t]);
      }
    }
#pragma unroll
    for (int t = 0; t < 4; ++t) {
      float bias = b2[a * NH2 + t * 16 + nl];
#pragma unroll
      for (int v = 0; v < 8; ++v) {
        int m = v + half * 8;
        float val = fmaxf(acc[t][v] + bias, 0.0f);
        sH2[(m0 + m) * 64 + t * 16 + nl] = (__bf16)val;
      }
    }
  }
  __syncthreads();

  // ---- layer 3: q = h2 @ W3 + b3 : 16 x 16 (8 valid cols) ----
  {
    v8f acc = {};
#pragma unroll
    for (int kk = 0; kk < 64; kk += 32) {
      v16bf af = lds_frag(&sH2[(m0 + nl) * 64 + kk], half);
      v16bf bf = lds_frag(&sW3t[nl * 64 + kk], half);
      acc = wmma_bf16(af, bf, acc);
    }
    float bias = (nl < NACT) ? b3[a * NACT + nl] : -1e30f;
#pragma unroll
    for (int v = 0; v < 8; ++v) {
      int m = v + half * 8;
      sQ[(m0 + m) * 16 + nl] = acc[v] + bias;
    }
  }
  __syncthreads();

  // ---- row max / argmax ----
  if (tid < 128) {
    float best = sQ[tid * 16 + 0];
    int   bi   = 0;
#pragma unroll
    for (int c = 1; c < NACT; ++c) {
      float q = sQ[tid * 16 + c];
      if (q > best) { best = q; bi = c; }
    }
    qs[(size_t)(b0 + tid) * NA + a]      = best;
    actions[(size_t)(b0 + tid) * NA + a] = bi;
  }
}

// ---------------------------------------------------------------------------
// Kernel 2: hypernetwork GEMM  out = epi(X @ W + bias)
// X: [BSZ, K] row-major, W: [K, N] row-major.
// Block tile 128 rows x (NT*16) cols, 8 waves. The A slab (128x32 f32,
// 16 KB) is DMA'd by the Tensor Data Mover into double-buffered LDS so the
// fetch of slab i+1 overlaps the WMMA work on slab i (TENSORcnt-tracked);
// B slabs are transposed/converted to bf16 by the waves.
// MODE: 0 = identity, 1 = abs, 2 = relu.
// ---------------------------------------------------------------------------
template <int NT, int MODE>
__global__ __launch_bounds__(256) void hyper_gemm_kernel(
    const float* __restrict__ X, const float* __restrict__ W,
    const float* __restrict__ bias, float* __restrict__ out,
    int K, int N)
{
  __shared__ __align__(16) float  sAf[2][128 * 32];   // TDM-staged f32 slabs
  __shared__ __align__(16) __bf16 sB[128 * 32];       // NT*16 cols x 32 K (transposed)

  const int tid   = threadIdx.x;
  const int lane  = tid & 31;
  const int wv    = tid >> 5;
  const int half  = lane >> 4;
  const int nl    = lane & 15;
  const int rowB  = blockIdx.x * 128;
  const int colB  = blockIdx.y * (NT * 16);
  constexpr int NC = NT * 16;

  v8f acc[NT] = {};

  // Prime the pipeline: DMA slab 0. TDM ignores EXEC; one wave issues one op.
  if (wv == 0) {
    tdm_load_2d_f32((unsigned)(size_t)&sAf[0][0],
                    X + (size_t)rowB * K, 32u, 128u, (unsigned long)K);
  }

  for (int kk = 0; kk < K; kk += 32) {
    const int cur = (kk >> 5) & 1;
    // stage B slab transposed: coalesced along n, LDS as [n][k]
    for (int idx = tid; idx < 32 * NC; idx += 256) {
      int k = idx / NC;
      int n = idx - k * NC;
      sB[n * 32 + k] = (__bf16)W[(size_t)(kk + k) * N + colB + n];
    }
    if (wv == 0) {
      const bool more = (kk + 32) < K;
      if (more) {  // DMA next A slab into the other buffer (overlaps compute)
        tdm_load_2d_f32((unsigned)(size_t)&sAf[cur ^ 1][0],
                        X + (size_t)rowB * K + kk + 32, 32u, 128u,
                        (unsigned long)K);
        __builtin_amdgcn_s_wait_tensorcnt(1);   // current slab complete
      } else {
        __builtin_amdgcn_s_wait_tensorcnt(0);
      }
    }
    if (kk + 32 < K)   // hint next B slab into L2 (global_prefetch_b8)
      __builtin_prefetch(W + (size_t)(kk + 32) * N + colB + nl, 0, 1);
    __syncthreads();

    v16bf af = lds_frag_f32(&sAf[cur][(wv * 16 + nl) * 32], half);
#pragma unroll
    for (int t = 0; t < NT; ++t) {
      v16bf bf = lds_frag(&sB[(t * 16 + nl) * 32], half);
      acc[t] = wmma_bf16(af, bf, acc[t]);
    }
    __syncthreads();
  }

#pragma unroll
  for (int t = 0; t < NT; ++t) {
    float bv = bias[colB + t * 16 + nl];
#pragma unroll
    for (int v = 0; v < 8; ++v) {
      float val = acc[t][v] + bv;
      if (MODE == 1) val = fabsf(val);
      if (MODE == 2) val = fmaxf(val, 0.0f);
      int row = rowB + wv * 16 + v + half * 8;
      out[(size_t)row * N + colB + t * 16 + nl] = val;
    }
  }
}

// ---------------------------------------------------------------------------
// Kernel 3: fused mixing head. One wave per batch row; lane == e (NE==32).
// q_tot[b] = sum_e elu(sum_a qs[b,a]*w1abs[b,a*E+e] + b1m[b,e]) * wf[b,e]
//          + sum_e v1[b,e]*Wv2[e] + bv2
// ---------------------------------------------------------------------------
__global__ __launch_bounds__(256) void mix_reduce_kernel(
    const float* __restrict__ qs, const float* __restrict__ w1abs,
    const float* __restrict__ b1m, const float* __restrict__ wf,
    const float* __restrict__ v1, const float* __restrict__ Wv2,
    const float* __restrict__ bv2, float* __restrict__ qtot)
{
  const int lane = threadIdx.x & 31;
  const int b    = blockIdx.x * 8 + (threadIdx.x >> 5);

  const float* wrow = w1abs + (size_t)b * (NA * NE);
  const float* qrow = qs + (size_t)b * NA;

  float acc = 0.0f;
#pragma unroll 4
  for (int a = 0; a < NA; ++a)                  // lanes read 128B coalesced
    acc = fmaf(qrow[a], wrow[a * NE + lane], acc);

  float hidden = acc + b1m[(size_t)b * NE + lane];
  hidden = (hidden > 0.0f) ? hidden : (__expf(hidden) - 1.0f);   // elu
  float t = hidden * wf[(size_t)b * NE + lane]
          + v1[(size_t)b * NE + lane] * Wv2[lane];
#pragma unroll
  for (int off = 16; off > 0; off >>= 1) t += __shfl_xor(t, off, 32);
  if (lane == 0) qtot[b] = t + bv2[0];
}

// ---------------------------------------------------------------------------
extern "C" void kernel_launch(void* const* d_in, const int* in_sizes, int n_in,
                              void* d_out, int out_size, void* d_ws, size_t ws_size,
                              hipStream_t stream)
{
  (void)in_sizes; (void)n_in; (void)out_size; (void)ws_size;
  const float* obs    = (const float*)d_in[0];
  const float* states = (const float*)d_in[1];
  const float* W1  = (const float*)d_in[2];
  const float* b1  = (const float*)d_in[3];
  const float* W2  = (const float*)d_in[4];
  const float* b2  = (const float*)d_in[5];
  const float* W3  = (const float*)d_in[6];
  const float* b3  = (const float*)d_in[7];
  const float* Wh1 = (const float*)d_in[8];
  const float* bh1 = (const float*)d_in[9];
  const float* Whb = (const float*)d_in[10];
  const float* bhb = (const float*)d_in[11];
  const float* Whf = (const float*)d_in[12];
  const float* bhf = (const float*)d_in[13];
  const float* Wv1 = (const float*)d_in[14];
  const float* bv1 = (const float*)d_in[15];
  const float* Wv2 = (const float*)d_in[16];
  const float* bv2 = (const float*)d_in[17];

  // Outputs: q_tot [BSZ] floats, then actions [BSZ*NA] as int32 bit patterns.
  float* qtot   = (float*)d_out;
  int* actions  = ((int*)d_out) + BSZ;

  // Workspace layout (floats): qs | w1abs | b1m | wf | v1  (~71 MB)
  float* ws    = (float*)d_ws;
  float* qs    = ws;
  float* w1abs = qs    + (size_t)BSZ * NA;
  float* b1m   = w1abs + (size_t)BSZ * NA * NE;
  float* wfin  = b1m   + (size_t)BSZ * NE;
  float* v1s   = wfin  + (size_t)BSZ * NE;

  agent_mlp_kernel<<<dim3(BSZ / 128, NA), 256, 0, stream>>>(
      obs, W1, b1, W2, b2, W3, b3, qs, actions);

  hyper_gemm_kernel<8, 1><<<dim3(BSZ / 128, (NA * NE) / 128), 256, 0, stream>>>(
      states, Wh1, bh1, w1abs, SD, NA * NE);
  hyper_gemm_kernel<2, 0><<<dim3(BSZ / 128, 1), 256, 0, stream>>>(
      states, Whb, bhb, b1m, SD, NE);
  hyper_gemm_kernel<2, 1><<<dim3(BSZ / 128, 1), 256, 0, stream>>>(
      states, Whf, bhf, wfin, SD, NE);
  hyper_gemm_kernel<2, 2><<<dim3(BSZ / 128, 1), 256, 0, stream>>>(
      states, Wv1, bv1, v1s, SD, NE);

  mix_reduce_kernel<<<BSZ / 8, 256, 0, stream>>>(
      qs, w1abs, b1m, wfin, v1s, Wv2, bv2, qtot);
}
